// T5EncoderWrapper_52312701665469
// MI455X (gfx1250) — matrix operand order repaired and checked by reading, not compiled, not run on hardware
//
#include <hip/hip_runtime.h>
#include <hip/hip_bf16.h>

// ---------------- types ----------------
typedef __attribute__((ext_vector_type(16))) __bf16 v16bf;
typedef __attribute__((ext_vector_type(8)))  __bf16 v8bf;
typedef __attribute__((ext_vector_type(8)))  float  v8f;
typedef __attribute__((ext_vector_type(4)))  unsigned int v4u;
typedef __attribute__((ext_vector_type(8)))  int v8i;
typedef __attribute__((ext_vector_type(4)))  int v4i;

#define T5_B   4
#define T5_S   2048
#define T5_D   512
#define T5_H   8
#define T5_DK  64
#define T5_DFF 2048
#define T5_L   6
#define T5_M   (T5_B * T5_S)   // 8192 rows

#if __has_builtin(__builtin_amdgcn_tensor_load_to_lds) && \
    __has_builtin(__builtin_amdgcn_s_wait_tensorcnt)
#define T5_HAVE_TDM 1
#else
#define T5_HAVE_TDM 0
#endif

__device__ __forceinline__ __bf16 f2bf(float f) {
    unsigned u = __builtin_bit_cast(unsigned, f);
    u += 0x7FFFu + ((u >> 16) & 1u);          // round-to-nearest-even
    unsigned short s = (unsigned short)(u >> 16);
    return __builtin_bit_cast(__bf16, s);
}

// A-fragment (16x32 bf16, M = lane&15):
// lanes 0-15 : K = [0..7] and [16..23] ; lanes 16-31 : K = [8..15] and [24..31]
__device__ __forceinline__ v16bf load_a_frag(const __bf16* A, int lda, int row0,
                                             int k0, int lane) {
    int m  = row0 + (lane & 15);
    int lo = (lane >> 4) * 8;
    const __bf16* p = A + (size_t)m * lda + k0 + lo;
    v8bf x = *(const v8bf*)p;
    v8bf y = *(const v8bf*)(p + 16);
    v16bf r;
#pragma unroll
    for (int i = 0; i < 8; ++i) { r[i] = x[i]; r[8 + i] = y[i]; }
    return r;
}

// B-fragment (32x16 bf16, from transposed Bt[N][K]):
// N = lane&15 ; lanes 0-15 hold K=[0..7] and [16..23], lanes 16-31 hold K=[8..15] and [24..31]
__device__ __forceinline__ v16bf load_b_frag(const __bf16* Bt, int ldb, int col0,
                                             int k0, int lane) {
    int n  = col0 + (lane & 15);
    int ko = (lane >> 4) * 8;
    const __bf16* p = Bt + (size_t)n * ldb + k0 + ko;
    v8bf x = *(const v8bf*)p;
    v8bf y = *(const v8bf*)(p + 16);
    v16bf r;
#pragma unroll
    for (int i = 0; i < 8; ++i) { r[i] = x[i]; r[8 + i] = y[i]; }
    return r;
}

__device__ __forceinline__ v8f wmma_bf16(v16bf a, v16bf b, v8f c) {
    return __builtin_amdgcn_wmma_f32_16x16x32_bf16(false, a, false, b,
                                                   (short)0, c, false, false);
}

// ---------------- TDM: DMA one [rows x 32bf16] tile into LDS ----------------
// Tensor = Bt-style matrix rows of K bf16 (row stride K*2 bytes).
// Tile rows are 64B; pad_interval=64B, pad_amount=16B -> LDS row stride 80B.
#define LDSROW 40   // bf16 elements per padded LDS row (80 bytes)

__device__ __forceinline__ void tdm_load_tile(const __bf16* g, unsigned lds_off,
                                              int rows, int K) {
#if T5_HAVE_TDM
    unsigned long long ga = (unsigned long long)(uintptr_t)g;
    int td0 = K >> 2;                 // row length in 8-byte units
    v4u g0;
    g0[0] = 1u;                                            // count=1
    g0[1] = lds_off;                                       // lds_addr (bytes)
    g0[2] = (unsigned)ga;                                  // global_addr lo
    g0[3] = (unsigned)((ga >> 32) & 0x01FFFFFFu) | (2u << 30);  // addr hi | type=2
    v8i g1;
    g1[0] = (3 << 16) | (1 << 20) | (3 << 22) | (3 << 25); // 8B elems, pad 16B/64B
    g1[1] = (td0 & 0xFFFF) << 16;                          // tensor_dim0 lo16
    g1[2] = ((td0 >> 16) & 0xFFFF) | ((rows & 0xFFFF) << 16); // dim0 hi | dim1 lo
    g1[3] = ((rows >> 16) & 0xFFFF) | (8 << 16);           // dim1 hi | tile_dim0=8
    g1[4] = rows & 0xFFFF;                                 // tile_dim1 (tile_dim2=0)
    g1[5] = td0;                                           // dim0_stride lo32
    g1[6] = 0;
    g1[7] = 0;
    v4i z4 = {0, 0, 0, 0};
    v8i z8 = {0, 0, 0, 0, 0, 0, 0, 0};
    __builtin_amdgcn_tensor_load_to_lds(g0, g1, z4, z4, z8, 0);
#else
    (void)g; (void)lds_off; (void)rows; (void)K;
#endif
}

// fragment loads out of the padded LDS tiles
__device__ __forceinline__ v16bf lds_a_frag(const __bf16* base, int row0, int lane) {
    int m  = row0 + (lane & 15);
    int lo = (lane >> 4) * 8;
    const __bf16* p = base + m * LDSROW + lo;
    v8bf x = *(const v8bf*)p;
    v8bf y = *(const v8bf*)(p + 16);
    v16bf r;
#pragma unroll
    for (int i = 0; i < 8; ++i) { r[i] = x[i]; r[8 + i] = y[i]; }
    return r;
}
__device__ __forceinline__ v16bf lds_b_frag(const __bf16* base, int col0, int lane) {
    int n  = col0 + (lane & 15);
    int ko = (lane >> 4) * 8;
    const __bf16* p = base + n * LDSROW + ko;
    v8bf x = *(const v8bf*)p;
    v8bf y = *(const v8bf*)(p + 16);
    v16bf r;
#pragma unroll
    for (int i = 0; i < 8; ++i) { r[i] = x[i]; r[8 + i] = y[i]; }
    return r;
}

// ---------------- embedding gather ----------------
__global__ void t5enc_embed(const int* __restrict__ ids,
                            const float* __restrict__ emb,
                            float* __restrict__ h) {
    int m  = blockIdx.x;
    int id = ids[m];
    for (int d = threadIdx.x; d < T5_D; d += blockDim.x)
        h[(size_t)m * T5_D + d] = emb[(size_t)id * T5_D + d];
}

// ---------------- RMSNorm (one wave32 per row) ----------------
template <int OUTF32>
__global__ void t5enc_rmsnorm(const float* __restrict__ hin,
                              const float* __restrict__ w,
                              void* __restrict__ out) {
    int row  = blockIdx.x;
    int lane = threadIdx.x;
    const float* p = hin + (size_t)row * T5_D;
    float vals[16];
    float ss = 0.f;
#pragma unroll
    for (int i = 0; i < 16; ++i) {
        float v = p[i * 32 + lane];
        vals[i] = v;
        ss += v * v;
    }
#pragma unroll
    for (int m = 16; m >= 1; m >>= 1) ss += __shfl_xor(ss, m, 32);
    float sc = rsqrtf(ss * (1.0f / T5_D) + 1e-6f);
    if (OUTF32) {
        float* o = (float*)out + (size_t)row * T5_D;
#pragma unroll
        for (int i = 0; i < 16; ++i)
            o[i * 32 + lane] = vals[i] * sc * w[i * 32 + lane];
    } else {
        __bf16* o = (__bf16*)out + (size_t)row * T5_D;
#pragma unroll
        for (int i = 0; i < 16; ++i)
            o[i * 32 + lane] = f2bf(vals[i] * sc * w[i * 32 + lane]);
    }
}

// ---------------- weight convert + transpose: f32[K,N] -> bf16[N,K] ----------------
__global__ void t5enc_wtrans(const float* __restrict__ W, __bf16* __restrict__ Wt,
                             int K, int N) {
    size_t idx = (size_t)blockIdx.x * blockDim.x + threadIdx.x;
    if (idx < (size_t)K * N) {
        int k = (int)(idx / N), n = (int)(idx % N);
        Wt[(size_t)n * K + k] = f2bf(W[idx]);
    }
}

// ---------------- bf16 WMMA GEMM with TDM-staged LDS tiles ----------------
// C[M,N] = A[M,K] * Bt[N,K]^T ; block = 8 waves = 2(M) x 4(N); block tile 64x256,
// wave tile 32x64. A/B tiles double-buffered in LDS via tensor_load_to_lds.
// MODE 0: out bf16[M,N]  1: relu->bf16[M,N]  2: f32 out[M,N] += C  3: bf16 out[N,M]
#define GEMM_BM 64
#define GEMM_BN 256

template <int MODE>
__global__ __launch_bounds__(256) void t5enc_gemm(const __bf16* __restrict__ A,
                                                  const __bf16* __restrict__ Bt,
                                                  void* __restrict__ out,
                                                  int K, int ldo) {
    __shared__ __align__(16) __bf16 Bs[2][GEMM_BN * LDSROW];
    __shared__ __align__(16) __bf16 As[2][GEMM_BM * LDSROW];
    int lane = threadIdx.x & 31;
    int wid  = threadIdx.x >> 5;
    int wm = wid & 1, wn = wid >> 1;               // 2 x 4 waves
    int tmb = blockIdx.x * GEMM_BM;
    int tnb = blockIdx.y * GEMM_BN;

    v8f c[2][4] = {};

#if T5_HAVE_TDM
    unsigned offB[2] = { (unsigned)(uintptr_t)&Bs[0][0], (unsigned)(uintptr_t)&Bs[1][0] };
    unsigned offA[2] = { (unsigned)(uintptr_t)&As[0][0], (unsigned)(uintptr_t)&As[1][0] };
    if (wid == 0)      tdm_load_tile(Bt + (size_t)tnb * K, offB[0], GEMM_BN, K);
    else if (wid == 1) tdm_load_tile(A  + (size_t)tmb * K, offA[0], GEMM_BM, K);
    for (int k0 = 0; k0 < K; k0 += 32) {
        int cur = (k0 >> 5) & 1;
        int k1 = k0 + 32;
        if (k1 < K) {
            if (wid == 0)      tdm_load_tile(Bt + (size_t)tnb * K + k1, offB[cur ^ 1], GEMM_BN, K);
            else if (wid == 1) tdm_load_tile(A  + (size_t)tmb * K + k1, offA[cur ^ 1], GEMM_BM, K);
            if (wid < 2) __builtin_amdgcn_s_wait_tensorcnt(1);
        } else {
            if (wid < 2) __builtin_amdgcn_s_wait_tensorcnt(0);
        }
        __syncthreads();                    // staged tiles visible to all waves
        v16bf a0 = lds_a_frag(As[cur], wm * 32,      lane);
        v16bf a1 = lds_a_frag(As[cur], wm * 32 + 16, lane);
#pragma unroll
        for (int j = 0; j < 4; ++j) {
            v16bf b = lds_b_frag(Bs[cur], wn * 64 + j * 16, lane);
            c[0][j] = wmma_bf16(a0, b, c[0][j]);
            c[1][j] = wmma_bf16(a1, b, c[1][j]);
        }
        __syncthreads();                    // all reads done before buffer reuse
    }
#else
    // fallback: cooperative single-buffer staging with plain loads
    for (int k0 = 0; k0 < K; k0 += 32) {
        int tid = threadIdx.x;
        {
            const __bf16* g = Bt + (size_t)(tnb + tid) * K + k0;
            v8bf* d = (v8bf*)&Bs[0][tid * LDSROW];
            d[0] = ((const v8bf*)g)[0]; d[1] = ((const v8bf*)g)[1];
            d[2] = ((const v8bf*)g)[2]; d[3] = ((const v8bf*)g)[3];
        }
        if (tid < GEMM_BM) {
            const __bf16* g = A + (size_t)(tmb + tid) * K + k0;
            v8bf* d = (v8bf*)&As[0][tid * LDSROW];
            d[0] = ((const v8bf*)g)[0]; d[1] = ((const v8bf*)g)[1];
            d[2] = ((const v8bf*)g)[2]; d[3] = ((const v8bf*)g)[3];
        }
        __syncthreads();
        v16bf a0 = lds_a_frag(As[0], wm * 32,      lane);
        v16bf a1 = lds_a_frag(As[0], wm * 32 + 16, lane);
#pragma unroll
        for (int j = 0; j < 4; ++j) {
            v16bf b = lds_b_frag(Bs[0], wn * 64 + j * 16, lane);
            c[0][j] = wmma_bf16(a0, b, c[0][j]);
            c[1][j] = wmma_bf16(a1, b, c[1][j]);
        }
        __syncthreads();
    }
#endif

    int half = lane >> 4, nl = lane & 15;
#pragma unroll
    for (int i = 0; i < 2; ++i)
#pragma unroll
        for (int j = 0; j < 4; ++j)
#pragma unroll
            for (int r = 0; r < 8; ++r) {
                int m = tmb + wm * 32 + i * 16 + r + half * 8;
                int n = tnb + wn * 64 + j * 16 + nl;
                float v = c[i][j][r];
                if (MODE == 0)
                    ((__bf16*)out)[(size_t)m * ldo + n] = f2bf(v);
                else if (MODE == 1)
                    ((__bf16*)out)[(size_t)m * ldo + n] = f2bf(v > 0.f ? v : 0.f);
                else if (MODE == 2)
                    ((float*)out)[(size_t)m * ldo + n] += v;
                else
                    ((__bf16*)out)[(size_t)n * ldo + m] = f2bf(v);
            }
}

// ---------------- flash attention ----------------
// grid = B*H*(S/64), block = 128 (4 waves, each wave owns 16 q rows)
// Q,K : bf16[B*S, D] (head slice at col h*64) ; Vt : bf16[D, B*S] (row = h*64+d)
__global__ __launch_bounds__(128) void t5enc_attn(const __bf16* __restrict__ Q,
                                                  const __bf16* __restrict__ Kb,
                                                  const __bf16* __restrict__ Vt,
                                                  const float* __restrict__ bias,
                                                  const float* __restrict__ mask,
                                                  __bf16* __restrict__ ctx) {
    int blk = blockIdx.x;
    int qt  = blk & 31;            // S/64 = 32 q-tiles
    int h   = (blk >> 5) & 7;
    int b   = blk >> 8;
    int wid  = threadIdx.x >> 5;
    int lane = threadIdx.x & 31;
    int half = lane >> 4, nl = lane & 15;
    int qb = qt * 64 + wid * 16;                 // wave's q-row base (within seq)

    __shared__ __align__(16) __bf16 Pl[4][16][32];

    // Q fragments (K-dim = DK = 64 -> two 32-wide steps)
    v16bf aq0 = load_a_frag(Q, T5_D, b * T5_S + qb, h * T5_DK + 0,  lane);
    v16bf aq1 = load_a_frag(Q, T5_D, b * T5_S + qb, h * T5_DK + 32, lane);

    float rmax[8], rsum[8];
#pragma unroll
    for (int r = 0; r < 8; ++r) { rmax[r] = -3.0e38f; rsum[r] = 0.f; }
    v8f acc[4] = {};

    for (int k0 = 0; k0 < T5_S; k0 += 32) {
        v8f s[2];
#pragma unroll
        for (int nt = 0; nt < 2; ++nt) {
            int kcb = b * T5_S + k0 + nt * 16;   // key rows (as B-matrix N)
            v16bf bk0 = load_b_frag(Kb, T5_D, kcb, h * T5_DK + 0,  lane);
            v16bf bk1 = load_b_frag(Kb, T5_D, kcb, h * T5_DK + 32, lane);
            v8f cf = {};
            cf = wmma_bf16(aq0, bk0, cf);
            cf = wmma_bf16(aq1, bk1, cf);
            int kcol = k0 + nt * 16 + nl;
            float em = (1.0f - mask[b * T5_S + kcol]) * -3.4028234663852886e38f;
#pragma unroll
            for (int r = 0; r < 8; ++r) {
                int q = qb + r + half * 8;
                cf[r] += bias[((size_t)h * T5_S + q) * T5_S + kcol] + em;
            }
            s[nt] = cf;
        }
        // online softmax (row stats reduced over 16-lane halves)
        float scl[8];
#pragma unroll
        for (int r = 0; r < 8; ++r) {
            float tm = fmaxf(s[0][r], s[1][r]);
            tm = fmaxf(tm, __shfl_xor(tm, 1, 32));
            tm = fmaxf(tm, __shfl_xor(tm, 2, 32));
            tm = fmaxf(tm, __shfl_xor(tm, 4, 32));
            tm = fmaxf(tm, __shfl_xor(tm, 8, 32));
            float nm = fmaxf(rmax[r], tm);
            float sc = __expf(rmax[r] - nm);
            rmax[r] = nm;
            float p0 = __expf(s[0][r] - nm);
            float p1 = __expf(s[1][r] - nm);
            s[0][r] = p0; s[1][r] = p1;
            float ps = p0 + p1;
            ps += __shfl_xor(ps, 1, 32);
            ps += __shfl_xor(ps, 2, 32);
            ps += __shfl_xor(ps, 4, 32);
            ps += __shfl_xor(ps, 8, 32);
            rsum[r] = rsum[r] * sc + ps;
            scl[r] = sc;
        }
#pragma unroll
        for (int t = 0; t < 4; ++t)
#pragma unroll
            for (int r = 0; r < 8; ++r) acc[t][r] *= scl[r];
        // stage P (exp scores) into LDS to re-layout as an A fragment
#pragma unroll
        for (int nt = 0; nt < 2; ++nt)
#pragma unroll
            for (int r = 0; r < 8; ++r)
                Pl[wid][r + half * 8][nt * 16 + nl] = f2bf(s[nt][r]);
        __syncthreads();
        v16bf ap = load_a_frag(&Pl[wid][0][0], 32, 0, 0, lane);
#pragma unroll
        for (int t = 0; t < 4; ++t) {
            v16bf bv = load_b_frag(Vt, T5_M, h * T5_DK + t * 16,
                                   b * T5_S + k0, lane);
            acc[t] = wmma_bf16(ap, bv, acc[t]);
        }
        __syncthreads();
    }
    // normalize + store ctx bf16 [B*S, D]
#pragma unroll
    for (int r = 0; r < 8; ++r) rsum[r] = 1.0f / rsum[r];
#pragma unroll
    for (int t = 0; t < 4; ++t)
#pragma unroll
        for (int r = 0; r < 8; ++r) {
            int q = qb + r + half * 8;
            int d = t * 16 + nl;
            ctx[((size_t)(b * T5_S) + q) * T5_D + h * T5_DK + d] =
                f2bf(acc[t][r] * rsum[r]);
        }
}

// ---------------- launcher ----------------
extern "C" void kernel_launch(void* const* d_in, const int* in_sizes, int n_in,
                              void* d_out, int out_size, void* d_ws, size_t ws_size,
                              hipStream_t stream) {
    (void)in_sizes; (void)n_in; (void)out_size; (void)ws_size;
    const int*   ids  = (const int*)d_in[0];
    const float* mask = (const float*)d_in[1];
    const float* bias = (const float*)d_in[2];
    const float* emb  = (const float*)d_in[3];
    const float* ln1  = (const float*)d_in[4];
    const float* Wq   = (const float*)d_in[5];
    const float* Wk   = (const float*)d_in[6];
    const float* Wv   = (const float*)d_in[7];
    const float* Wo   = (const float*)d_in[8];
    const float* ln2  = (const float*)d_in[9];
    const float* Wi   = (const float*)d_in[10];
    const float* Woff = (const float*)d_in[11];
    const float* lnf  = (const float*)d_in[12];

    char* ws = (char*)d_ws;
    size_t off = 0;
    auto alloc = [&](size_t bytes) {
        size_t p = off;
        off = (off + bytes + 255) & ~(size_t)255;
        return p;
    };
    float*  h    = (float*)(ws + alloc((size_t)T5_M * T5_D * 4));
    __bf16* x    = (__bf16*)(ws + alloc((size_t)T5_M * T5_D * 2));
    __bf16* qb   = (__bf16*)(ws + alloc((size_t)T5_M * T5_D * 2));
    __bf16* kb   = (__bf16*)(ws + alloc((size_t)T5_M * T5_D * 2));
    __bf16* vt   = (__bf16*)(ws + alloc((size_t)T5_M * T5_D * 2));
    __bf16* ctxb = (__bf16*)(ws + alloc((size_t)T5_M * T5_D * 2));
    __bf16* act  = (__bf16*)(ws + alloc((size_t)T5_M * T5_DFF * 2));
    __bf16* wt   = (__bf16*)(ws + alloc((size_t)T5_L * 3145728 * 2));

    // per-layer bf16 transposed weights: [WqT WkT WvT WoT | WiT WoffT]
    const size_t LW = 3145728;                  // elements per layer
    for (int l = 0; l < T5_L; ++l) {
        __bf16* wl = wt + (size_t)l * LW;
        t5enc_wtrans<<<1024, 256, 0, stream>>>(Wq + (size_t)l * 262144, wl + 0,       512, 512);
        t5enc_wtrans<<<1024, 256, 0, stream>>>(Wk + (size_t)l * 262144, wl + 262144,  512, 512);
        t5enc_wtrans<<<1024, 256, 0, stream>>>(Wv + (size_t)l * 262144, wl + 524288,  512, 512);
        t5enc_wtrans<<<1024, 256, 0, stream>>>(Wo + (size_t)l * 262144, wl + 786432,  512, 512);
        t5enc_wtrans<<<4096, 256, 0, stream>>>(Wi + (size_t)l * 1048576, wl + 1048576, 512, 2048);
        t5enc_wtrans<<<4096, 256, 0, stream>>>(Woff + (size_t)l * 1048576, wl + 2097152, 2048, 512);
    }

    t5enc_embed<<<T5_M, 256, 0, stream>>>(ids, emb, h);

    dim3 blk(256);
    dim3 gD(T5_M / GEMM_BM, T5_D / GEMM_BN);     // (128, 2)
    dim3 gF(T5_M / GEMM_BM, T5_DFF / GEMM_BN);   // (128, 8)

    for (int l = 0; l < T5_L; ++l) {
        __bf16* wqt = wt + (size_t)l * LW;
        __bf16* wkt = wqt + 262144;
        __bf16* wvt = wqt + 524288;
        __bf16* wot = wqt + 786432;
        __bf16* wit = wqt + 1048576;
        __bf16* wft = wqt + 2097152;

        t5enc_rmsnorm<0><<<T5_M, 32, 0, stream>>>(h, ln1 + l * T5_D, x);
        t5enc_gemm<0><<<gD, blk, 0, stream>>>(x, wqt, qb, T5_D, T5_D);
        t5enc_gemm<0><<<gD, blk, 0, stream>>>(x, wkt, kb, T5_D, T5_D);
        t5enc_gemm<3><<<gD, blk, 0, stream>>>(x, wvt, vt, T5_D, T5_M);   // transposed V
        t5enc_attn<<<T5_B * T5_H * (T5_S / 64), 128, 0, stream>>>(
            qb, kb, vt, bias, mask, ctxb);
        t5enc_gemm<2><<<gD, blk, 0, stream>>>(ctxb, wot, h, T5_D, T5_D); // h += ctx@Wo

        t5enc_rmsnorm<0><<<T5_M, 32, 0, stream>>>(h, ln2 + l * T5_D, x);
        t5enc_gemm<1><<<gF, blk, 0, stream>>>(x, wit, act, T5_D, T5_DFF);   // relu
        t5enc_gemm<2><<<gD, blk, 0, stream>>>(act, wft, h, T5_DFF, T5_D);   // h += ffn
    }

    t5enc_rmsnorm<1><<<T5_M, 32, 0, stream>>>(h, lnf, d_out);
}
